// Bank_selfAtt_5488968204306
// MI455X (gfx1250) — matrix operand order repaired and checked by reading, not compiled, not run on hardware
//
#include <hip/hip_runtime.h>

// ---------------------------------------------------------------------------
// Types for CDNA5 WMMA (wave32): v_wmma_f32_16x16x32_bf16
// ---------------------------------------------------------------------------
typedef __attribute__((ext_vector_type(16))) __bf16 v16bf;
typedef __attribute__((ext_vector_type(8)))  float  v8f;
typedef __attribute__((ext_vector_type(8)))  unsigned int v8u;
typedef __attribute__((ext_vector_type(4)))  unsigned int u32x4;
typedef __attribute__((ext_vector_type(8)))  int i32x8;
typedef __attribute__((ext_vector_type(4)))  int i32x4;

#if defined(__has_builtin)
#if __has_builtin(__builtin_amdgcn_tensor_load_to_lds)
#define HAVE_TDM 1
#endif
#endif

__device__ inline v8f vzero() {
  v8f z = {0.f, 0.f, 0.f, 0.f, 0.f, 0.f, 0.f, 0.f};
  return z;
}

// float -> bf16 (round to nearest even)
__device__ inline unsigned short f2bf(float f) {
  unsigned int u = __float_as_uint(f);
  u += 0x7FFFu + ((u >> 16) & 1u);
  return (unsigned short)(u >> 16);
}

// Build a 16-element bf16 fragment from two 16-byte chunks (ISA A-layout: the
// per-lane 16 values are two contiguous 8-value runs).
__device__ inline v16bf frag_2x16(const unsigned short* p0, const unsigned short* p1) {
  v8u r;
  uint4 a = *(const uint4*)p0;
  uint4 b = *(const uint4*)p1;
  r[0] = a.x; r[1] = a.y; r[2] = a.z; r[3] = a.w;
  r[4] = b.x; r[5] = b.y; r[6] = b.z; r[7] = b.w;
  return __builtin_bit_cast(v16bf, r);
}
// 32-byte contiguous fragment (16 contiguous K per lane-half, not the ISA table order)
__device__ inline v16bf frag_32B(const unsigned short* p) { return frag_2x16(p, p + 8); }

__device__ inline v8f wmma_bf16(v16bf a, v16bf b, v8f c) {
  // (neg_a, A, neg_b, B, c_mod, C, reuse_a, reuse_b)
  return __builtin_amdgcn_wmma_f32_16x16x32_bf16(false, a, false, b, (short)0, c, false, false);
}

// ---------------------------------------------------------------------------
// Problem constants
// ---------------------------------------------------------------------------
#define BB 8192
#define SS 128
#define HH 512

// ---------------------------------------------------------------------------
// Kernel 1: inner projection  h[b, 4s+j] = x[b,s]*iw[j] + ib[j]   (+ bf16 copy)
// ---------------------------------------------------------------------------
__global__ __launch_bounds__(256) void k_proj(const float* __restrict__ x,
                                              const float* __restrict__ iw,
                                              const float* __restrict__ ib,
                                              float* __restrict__ h,
                                              unsigned short* __restrict__ hbf) {
  int idx = blockIdx.x * 256 + threadIdx.x;   // [0, B*H)
  int b = idx >> 9, c = idx & 511;
  float v = x[(b << 7) + (c >> 2)] * iw[c & 3] + ib[c & 3];
  h[idx] = v;
  hbf[idx] = f2bf(v);
}

// ---------------------------------------------------------------------------
// Kernel 2: transpose + bf16-convert a 512x512 weight:  wt[n*512+k] = w[k*512+n]
// ---------------------------------------------------------------------------
__global__ __launch_bounds__(256) void k_prepw(const float* __restrict__ w,
                                               unsigned short* __restrict__ wt) {
  int idx = blockIdx.x * 256 + threadIdx.x;   // [0, 512*512)
  int k = idx >> 9, n = idx & 511;
  wt[(n << 9) + k] = f2bf(w[idx]);
}

// ---------------------------------------------------------------------------
// Kernel 3: WMMA GEMM  out = hbf @ W + bias   (W given transposed bf16)
// grid (B/64, 512/64), 128 threads (4 waves); wave w owns rows [64bx+16w, +16)
// vtrans==1 -> store transposed (for V): out[col*8192 + row]
// ---------------------------------------------------------------------------
__global__ __launch_bounds__(128) void k_qkv(const unsigned short* __restrict__ hbf,
                                             const unsigned short* __restrict__ wt,
                                             const float* __restrict__ bias,
                                             unsigned short* __restrict__ out,
                                             int vtrans) {
  const int tid = threadIdx.x, wid = tid >> 5, lane = tid & 31;
  const int half = lane >> 4, ln = lane & 15;
  const int rbase = blockIdx.x * 64 + wid * 16;
  const int cbase = blockIdx.y * 64;

  v8f acc[4];
  #pragma unroll
  for (int i = 0; i < 4; i++) acc[i] = vzero();

  const unsigned short* arow = hbf + (rbase + ln) * HH;
  for (int k0 = 0; k0 < HH; k0 += 32) {
    v16bf a = frag_2x16(arow + k0 + 8 * half, arow + k0 + 16 + 8 * half);
    #pragma unroll
    for (int ns = 0; ns < 4; ns++) {
      int col = cbase + ns * 16 + ln;
      v16bf b = frag_2x16(wt + (col << 9) + k0 + 8 * half, wt + (col << 9) + k0 + 16 + 8 * half);
      acc[ns] = wmma_bf16(a, b, acc[ns]);
    }
  }
  #pragma unroll
  for (int ns = 0; ns < 4; ns++) {
    int col = cbase + ns * 16 + ln;
    float bv = bias[col];
    #pragma unroll
    for (int e = 0; e < 8; e++) {
      int row = rbase + e + 8 * half;
      float val = acc[ns][e] + bv;
      if (vtrans) out[(col << 13) + row] = f2bf(val);
      else        out[(row << 9) + col] = f2bf(val);
    }
  }
}

// ---------------------------------------------------------------------------
// Kernel 4: flash attention across the batch axis (WMMA, online softmax)
// grid 256 blocks (32 Q rows each), 128 threads = 4 waves.
//   phase1: S = (Q Kt)/512  -- wave w computes 16 KV cols, both 16-row tiles
//   softmax: fully parallel online max/sum (4 threads per row)
//   phase2: O = O*rs + P V  -- wave w owns head cols [128w, 128w+128)
// Q tile staged into LDS by the Tensor Data Mover (padded rows via pad_enable);
// K/V stream from L2 with global_prefetch hints for the next tile.
// ---------------------------------------------------------------------------
#define QS_STRIDE 520   // 512 + pad (u16): TDM pad_interval=7(256dw), pad_amount=3(4dw)
#define SS_STRIDE 65    // floats
#define PS_STRIDE 72    // u16, 16B-aligned rows

__global__ __launch_bounds__(128, 1) void k_flash(const unsigned short* __restrict__ Qb,
                                                  const unsigned short* __restrict__ Kb,
                                                  const unsigned short* __restrict__ Vt,
                                                  float* __restrict__ O) {
  __shared__ __align__(16) unsigned short Qs[32 * QS_STRIDE];
  __shared__ __align__(16) float Ss[32 * SS_STRIDE];
  __shared__ __align__(16) unsigned short Ps[32 * PS_STRIDE];
  __shared__ float m_s[32], l_s[32], rs_s[32];
  __shared__ float red4[32][4], sum4[32][4];

  const int tid = threadIdx.x, wid = tid >> 5, lane = tid & 31;
  const int half = lane >> 4, ln = lane & 15;
  const int qbase = blockIdx.x * 32;

#ifdef HAVE_TDM
  // ---- TDM: each wave DMA's its 8 Q rows (16-bit 2D tile) into padded LDS ----
  {
    unsigned long long ga =
        (unsigned long long)(const void*)(Qb + (size_t)(qbase + wid * 8) * HH);
    unsigned lds_addr =
        (unsigned)(unsigned long long)(const void*)&Qs[wid * 8 * QS_STRIDE];
    u32x4 g0;
    g0[0] = 1u;                                    // count=1, user descriptor
    g0[1] = lds_addr;                              // lds_addr [63:32]
    g0[2] = (unsigned)(ga & 0xFFFFFFFFu);          // global_addr low
    g0[3] = (unsigned)((ga >> 32) & 0x01FFFFFFu)   // global_addr[56:32]
            | (2u << 30);                          // type=2 ("image")
    i32x8 g1;
    g1[0] = (int)((1u << 16)    // data_size = 2 bytes
                | (1u << 20)    // pad_enable
                | (7u << 22)    // pad_interval: every 256 DWORDs (=512 bf16)
                | (3u << 25));  // pad_amount: 4 DWORDs (=16B) -> row stride 520 u16
    g1[1] = (int)(512u << 16);  // tensor_dim0 = 512 (low16 @ bits 63:48)
    g1[2] = (int)(8192u << 16); // tensor_dim0 hi=0 | tensor_dim1 = 8192 (low16)
    g1[3] = (int)(512u << 16);  // tensor_dim1 hi=0 | tile_dim0 = 512
    g1[4] = 8;                  // tile_dim1 = 8 rows per wave, tile_dim2 = 0
    g1[5] = 512;                // tensor_dim0_stride low32
    g1[6] = 0;                  // stride0 hi | tensor_dim1_stride low16
    g1[7] = 0;
    i32x4 gz = {0, 0, 0, 0};
#if defined(__clang_major__) && (__clang_major__ >= 23)
    i32x8 gz8 = {0, 0, 0, 0, 0, 0, 0, 0};
    __builtin_amdgcn_tensor_load_to_lds(g0, g1, gz, gz, gz8, 0);
#else
    __builtin_amdgcn_tensor_load_to_lds(g0, g1, gz, gz, 0);
#endif
    __builtin_amdgcn_s_wait_tensorcnt(0);
  }
#else
  // fallback: cooperative copy of the Q tile into padded LDS rows
  for (int i = tid; i < 32 * 64; i += 128) {        // 64 uint4 per row
    int row = i >> 6, cb = i & 63;
    *((uint4*)(Qs + row * QS_STRIDE) + cb) =
        *((const uint4*)(Qb + (qbase + row) * HH) + cb);
  }
#endif
  if (tid < 32) { m_s[tid] = -3.0e38f; l_s[tid] = 0.f; }

  v8f acc[16];
  #pragma unroll
  for (int i = 0; i < 16; i++) acc[i] = vzero();
  __syncthreads();

  for (int kv0 = 0; kv0 < BB; kv0 += 64) {
    // ---- phase 1: S tile (32 x 64), wave handles 16 KV columns ----
    v8f s0 = vzero(), s1 = vzero();
    const unsigned short* krow = Kb + (kv0 + wid * 16 + ln) * HH;
    // prefetch next KV tile's K row for this lane (8 x 128B cachelines)
    if (kv0 + 64 < BB) {
      const unsigned short* knext = krow + 64 * HH;
      #pragma unroll
      for (int pf = 0; pf < 8; pf++) __builtin_prefetch(knext + pf * 64, 0, 3);
    }
    const unsigned short* q0r = Qs + ln * QS_STRIDE;
    const unsigned short* q1r = Qs + (16 + ln) * QS_STRIDE;
    for (int k0 = 0; k0 < HH; k0 += 32) {
      v16bf b  = frag_2x16(krow + k0 + 8 * half, krow + k0 + 16 + 8 * half);
      v16bf a0 = frag_2x16(q0r + k0 + 8 * half, q0r + k0 + 16 + 8 * half);
      s0 = wmma_bf16(a0, b, s0);
      v16bf a1 = frag_2x16(q1r + k0 + 8 * half, q1r + k0 + 16 + 8 * half);
      s1 = wmma_bf16(a1, b, s1);
    }
    const float sc = 1.0f / 512.0f;
    const int scol = wid * 16 + ln;
    #pragma unroll
    for (int e = 0; e < 8; e++) {
      Ss[(e + 8 * half) * SS_STRIDE + scol]      = s0[e] * sc;
      Ss[(16 + e + 8 * half) * SS_STRIDE + scol] = s1[e] * sc;
    }
    __syncthreads();

    // ---- online softmax: 4 threads per row, 16 columns each ----
    {
      const int r = tid >> 2, c4 = tid & 3;
      const float* srow = Ss + r * SS_STRIDE + c4 * 16;
      float lm = -3.0e38f;
      #pragma unroll
      for (int j = 0; j < 16; j++) lm = fmaxf(lm, srow[j]);
      red4[r][c4] = lm;
      __syncthreads();
      float mo = m_s[r];
      float mn = fmaxf(mo, fmaxf(fmaxf(red4[r][0], red4[r][1]),
                                 fmaxf(red4[r][2], red4[r][3])));
      float sum = 0.f;
      unsigned short* prow = Ps + r * PS_STRIDE + c4 * 16;
      #pragma unroll
      for (int j = 0; j < 16; j++) {
        float p = __expf(srow[j] - mn);
        sum += p;
        prow[j] = f2bf(p);
      }
      sum4[r][c4] = sum;
      __syncthreads();
      if (tid < 32) {
        float mo2 = m_s[tid];
        float mn2 = fmaxf(mo2, fmaxf(fmaxf(red4[tid][0], red4[tid][1]),
                                     fmaxf(red4[tid][2], red4[tid][3])));
        float rs = __expf(mo2 - mn2);
        l_s[tid] = l_s[tid] * rs +
                   (sum4[tid][0] + sum4[tid][1] + sum4[tid][2] + sum4[tid][3]);
        m_s[tid] = mn2;
        rs_s[tid] = rs;
      }
      __syncthreads();
    }

    // ---- phase 2: O = O * rs + P @ V  (wave owns 128 head cols) ----
    float rv0[8], rv1[8];
    #pragma unroll
    for (int e = 0; e < 8; e++) {
      rv0[e] = rs_s[e + 8 * half];
      rv1[e] = rs_s[16 + e + 8 * half];
    }
    #pragma unroll
    for (int t = 0; t < 8; t++)
      #pragma unroll
      for (int e = 0; e < 8; e++) { acc[t][e] *= rv0[e]; acc[8 + t][e] *= rv1[e]; }

    const unsigned short* p0r = Ps + ln * PS_STRIDE;
    const unsigned short* p1r = Ps + (16 + ln) * PS_STRIDE;
    #pragma unroll
    for (int k0 = 0; k0 < 64; k0 += 32) {
      v16bf a0 = frag_2x16(p0r + k0 + 8 * half, p0r + k0 + 16 + 8 * half);
      v16bf a1 = frag_2x16(p1r + k0 + 8 * half, p1r + k0 + 16 + 8 * half);
      #pragma unroll
      for (int ns = 0; ns < 8; ns++) {
        int col = wid * 128 + ns * 16 + ln;
        const unsigned short* vp = Vt + (col << 13) + kv0 + k0 + 8 * half;
        if (k0 == 0 && kv0 + 64 < BB)
          __builtin_prefetch(Vt + (col << 13) + kv0 + 64, 0, 3);  // next V slice
        v16bf b = frag_2x16(vp, vp + 16);
        acc[ns]     = wmma_bf16(a0, b, acc[ns]);
        acc[8 + ns] = wmma_bf16(a1, b, acc[8 + ns]);
      }
    }
    // no barrier needed here: next phase1 touches only Ss/Qs; Ps/rs_s rewritten
    // only after the post-phase1 barrier, which all waves reach after this point.
  }

  // ---- epilogue: normalize by l, store O (fp32) ----
  float lv0[8], lv1[8];
  #pragma unroll
  for (int e = 0; e < 8; e++) {
    lv0[e] = 1.0f / l_s[e + 8 * half];
    lv1[e] = 1.0f / l_s[16 + e + 8 * half];
  }
  #pragma unroll
  for (int ns = 0; ns < 8; ns++) {
    int col = wid * 128 + ns * 16 + ln;
    #pragma unroll
    for (int e = 0; e < 8; e++) {
      O[(qbase + e + 8 * half) * HH + col]      = acc[ns][e] * lv0[e];
      O[(qbase + 16 + e + 8 * half) * HH + col] = acc[8 + ns][e] * lv1[e];
    }
  }
}

// ---------------------------------------------------------------------------
// Kernel 5: residual + LN(512) + per-sample token attention + out proj + LN(128)
// one block (128 threads) per sample; thread t = token t
// ---------------------------------------------------------------------------
__device__ inline float bsum(float v, float* buf, int t) {
  buf[t] = v;
  __syncthreads();
  for (int s = 64; s > 0; s >>= 1) {
    if (t < s) buf[t] += buf[t + s];
    __syncthreads();
  }
  float r = buf[0];
  __syncthreads();
  return r;
}

__global__ __launch_bounds__(128) void k_stage2(
    const float* __restrict__ h, const float* __restrict__ Oacc,
    const float* __restrict__ ln1s, const float* __restrict__ ln1o,
    const float* __restrict__ wq1, const float* __restrict__ bq1,
    const float* __restrict__ wk1, const float* __restrict__ bk1,
    const float* __restrict__ wv1, const float* __restrict__ bv1,
    const float* __restrict__ outw, const float* __restrict__ outb,
    const float* __restrict__ ln2s, const float* __restrict__ ln2o,
    float* __restrict__ out) {
  __shared__ float red[128];
  __shared__ float K1s[128][8];
  __shared__ float V1s[128][4];
  const int t = threadIdx.x;
  const int b = blockIdx.x;
  const float* hb = h + b * HH;
  const float* ob = Oacc + b * HH;

  // residual + LN over H=512
  float x[4];
  float s = 0.f, s2 = 0.f;
  #pragma unroll
  for (int j = 0; j < 4; j++) {
    float v = hb[t * 4 + j] + ob[t * 4 + j];
    x[j] = v; s += v; s2 += v * v;
  }
  s = bsum(s, red, t);
  s2 = bsum(s2, red, t);
  float mu = s * (1.f / 512.f);
  float var = s2 * (1.f / 512.f) - mu * mu;
  float inv = rsqrtf(var + 1e-5f);
  float dp[4];
  #pragma unroll
  for (int j = 0; j < 4; j++)
    dp[j] = (x[j] - mu) * inv * ln1s[t * 4 + j] + ln1o[t * 4 + j];

  // tiny projections
  float q1[8];
  #pragma unroll
  for (int o = 0; o < 8; o++) {
    float a = bq1[o], k = bk1[o];
    #pragma unroll
    for (int j = 0; j < 4; j++) { a += dp[j] * wq1[j * 8 + o]; k += dp[j] * wk1[j * 8 + o]; }
    q1[o] = a; K1s[t][o] = k;
  }
  #pragma unroll
  for (int o = 0; o < 4; o++) {
    float v = bv1[o];
    #pragma unroll
    for (int j = 0; j < 4; j++) v += dp[j] * wv1[j * 4 + o];
    V1s[t][o] = v;
  }
  __syncthreads();

  // softmax attention over 128 tokens (two passes, recompute logits)
  float mx = -3.0e38f;
  for (int jj = 0; jj < 128; jj++) {
    float e = 0.f;
    #pragma unroll
    for (int o = 0; o < 8; o++) e += q1[o] * K1s[jj][o];
    mx = fmaxf(mx, e * 0.125f);
  }
  float sum = 0.f;
  float av[4] = {0.f, 0.f, 0.f, 0.f};
  for (int jj = 0; jj < 128; jj++) {
    float e = 0.f;
    #pragma unroll
    for (int o = 0; o < 8; o++) e += q1[o] * K1s[jj][o];
    float p = __expf(e * 0.125f - mx);
    sum += p;
    #pragma unroll
    for (int d = 0; d < 4; d++) av[d] += p * V1s[jj][d];
  }
  float oi = outb[0];
  float rinv = 1.f / sum;
  #pragma unroll
  for (int d = 0; d < 4; d++) oi += (av[d] * rinv + dp[d]) * outw[d];

  // LN over tokens
  float ts = bsum(oi, red, t);
  float ts2 = bsum(oi * oi, red, t);
  float mu2 = ts * (1.f / 128.f);
  float var2 = ts2 * (1.f / 128.f) - mu2 * mu2;
  float inv2 = rsqrtf(var2 + 1e-5f);
  out[b * SS + t] = (oi - mu2) * inv2 * ln2s[t] + ln2o[t];
}

// ---------------------------------------------------------------------------
// Launch
// ---------------------------------------------------------------------------
extern "C" void kernel_launch(void* const* d_in, const int* in_sizes, int n_in,
                              void* d_out, int out_size, void* d_ws, size_t ws_size,
                              hipStream_t stream) {
  (void)in_sizes; (void)n_in; (void)out_size; (void)ws_size;
  const float* x    = (const float*)d_in[0];
  const float* iw   = (const float*)d_in[1];
  const float* ib   = (const float*)d_in[2];
  const float* wq   = (const float*)d_in[3];
  const float* bq   = (const float*)d_in[4];
  const float* wk   = (const float*)d_in[5];
  const float* bk   = (const float*)d_in[6];
  const float* wv   = (const float*)d_in[7];
  const float* bv   = (const float*)d_in[8];
  const float* ln1s = (const float*)d_in[9];
  const float* ln1o = (const float*)d_in[10];
  const float* wq1  = (const float*)d_in[11];
  const float* bq1  = (const float*)d_in[12];
  const float* wk1  = (const float*)d_in[13];
  const float* bk1  = (const float*)d_in[14];
  const float* wv1  = (const float*)d_in[15];
  const float* bv1  = (const float*)d_in[16];
  const float* ow   = (const float*)d_in[17];
  const float* obb  = (const float*)d_in[18];
  const float* ln2s = (const float*)d_in[19];
  const float* ln2o = (const float*)d_in[20];

  char* ws = (char*)d_ws;
  float*          h   = (float*)(ws);                        // 0..16 MB
  unsigned short* hbf = (unsigned short*)(ws + (16u << 20)); // 16..24 MB
  unsigned short* Qb  = (unsigned short*)(ws + (24u << 20)); // 24..32 MB
  unsigned short* Kb  = (unsigned short*)(ws + (32u << 20)); // 32..40 MB
  unsigned short* Vt  = (unsigned short*)(ws + (40u << 20)); // 40..48 MB (transposed)
  float*          Oa  = (float*)(ws + (48u << 20));          // 48..64 MB
  unsigned short* WtQ = (unsigned short*)(ws + (64u << 20));
  unsigned short* WtK = WtQ + 512 * 512;
  unsigned short* WtV = WtK + 512 * 512;

  k_proj<<<(BB * HH) / 256, 256, 0, stream>>>(x, iw, ib, h, hbf);
  k_prepw<<<(512 * 512) / 256, 256, 0, stream>>>(wq, WtQ);
  k_prepw<<<(512 * 512) / 256, 256, 0, stream>>>(wk, WtK);
  k_prepw<<<(512 * 512) / 256, 256, 0, stream>>>(wv, WtV);

  dim3 g(BB / 64, HH / 64);
  k_qkv<<<g, 128, 0, stream>>>(hbf, WtQ, bq, Qb, 0);
  k_qkv<<<g, 128, 0, stream>>>(hbf, WtK, bk, Kb, 0);
  k_qkv<<<g, 128, 0, stream>>>(hbf, WtV, bv, Vt, 1);

  k_flash<<<BB / 32, 128, 0, stream>>>(Qb, Kb, Vt, Oa);

  k_stage2<<<BB, 128, 0, stream>>>(h, Oa, ln1s, ln1o, wq1, bq1, wk1, bk1,
                                   wv1, bv1, ow, obb, ln2s, ln2o, (float*)d_out);
}
